// SlotTextDecoderGRU_80968723464297
// MI455X (gfx1250) — matrix operand order repaired and checked by reading, not compiled, not run on hardware
//
#include <hip/hip_runtime.h>
#include <hip/hip_bf16.h>

// Problem constants (from the reference)
#define BATCH   8
#define SEQ     512
#define HID     1024
#define IN_DIM  768
#define VOCAB   32000
#define LAYERS  2

typedef __attribute__((ext_vector_type(16))) __bf16 v16bf;
typedef __attribute__((ext_vector_type(8)))  float  v8f;

union Frag16 { v16bf v; uint4 q[2]; };

#if defined(__has_builtin)
# if __has_builtin(__builtin_amdgcn_global_load_async_to_lds_b128)
#  define USE_ASYNC_LDS 1
# endif
# if __has_builtin(__builtin_amdgcn_s_wait_asynccnt)
#  define HAVE_WAIT_ASYNC_BUILTIN 1
# endif
#endif

// Exact parameter type per hipcc diagnostic:
//   __attribute__((vector_size(16))) int __device__ *   (AS1 source)
//   and the LDS destination in AS3.
typedef int v4i_gcc __attribute__((vector_size(16)));
typedef __attribute__((address_space(1))) v4i_gcc* as1_v4i;
typedef __attribute__((address_space(3))) v4i_gcc* as3_v4i;

__device__ __forceinline__ void stage16(const __bf16* g, __bf16* l) {
#if defined(USE_ASYNC_LDS)
  __builtin_amdgcn_global_load_async_to_lds_b128(
      (as1_v4i)(g), (as3_v4i)(l), /*imm offset=*/0, /*cpol=*/0);
#else
  *(uint4*)l = *(const uint4*)g;
#endif
}

__device__ __forceinline__ void wait_async0() {
#if defined(USE_ASYNC_LDS)
# if defined(HAVE_WAIT_ASYNC_BUILTIN)
  __builtin_amdgcn_s_wait_asynccnt(0);
# else
  asm volatile("s_wait_asynccnt 0x0" ::: "memory");
# endif
#endif
}

__device__ __forceinline__ v8f wmma_bf16(v16bf a, v16bf b, v8f c) {
  // D = A(16x32 bf16) * B(32x16 bf16) + C(16x16 f32)
  return __builtin_amdgcn_wmma_f32_16x16x32_bf16(
      /*neg_a=*/false, a, /*neg_b=*/false, b,
      /*c_mod=*/(short)0, c, /*reuse_a=*/false, /*reuse_b=*/false);
}

__device__ __forceinline__ float fast_sigmoid(float x) {
  return 1.0f / (1.0f + __expf(-x));
}
__device__ __forceinline__ float fast_tanh(float x) {
  float ax = fabsf(x);
  float e  = __expf(-2.0f * ax);          // in (0,1], no overflow
  float t  = (1.0f - e) / (1.0f + e);
  return copysignf(t, x);
}

// ---------------- elementwise / small kernels ----------------

__global__ void cvt_f32_bf16_kernel(const float* __restrict__ src,
                                    __bf16* __restrict__ dst, long n) {
  long i = (long)blockIdx.x * blockDim.x + threadIdx.x;
  if (i < n) dst[i] = (__bf16)src[i];
}

__global__ void gc_mean_kernel(const float* __restrict__ concepts,  // [B][16][768]
                               float* __restrict__ gc) {            // [B][768]
  int i = blockIdx.x * blockDim.x + threadIdx.x;
  if (i < BATCH * IN_DIM) {
    int b = i / IN_DIM, d = i - b * IN_DIM;
    float s = 0.f;
#pragma unroll
    for (int n = 0; n < 16; ++n)
      s += concepts[((long)b * 16 + n) * IN_DIM + d];
    gc[i] = s * (1.0f / 16.0f);
  }
}

__global__ void h0_kernel(const float* __restrict__ gc,    // [B][768]
                          const float* __restrict__ w,     // [2*HID][768]
                          const float* __restrict__ bias,  // [2*HID]
                          float* __restrict__ h0) {        // [L][B][HID]
  int i = blockIdx.x * blockDim.x + threadIdx.x;
  if (i < LAYERS * BATCH * HID) {
    int l = i >> 13;            // /8192
    int b = (i >> 10) & 7;
    int j = i & (HID - 1);
    int orow = l * HID + j;
    const float* grow = gc + b * IN_DIM;
    const float* wrow = w + (long)orow * IN_DIM;
    float s = bias[orow];
    for (int d = 0; d < IN_DIM; ++d) s += grow[d] * wrow[d];
    h0[i] = s;                  // layout: [l][b][j]
  }
}

__global__ void embed_gather_kernel(const float* __restrict__ emb,  // [VOCAB][HID]
                                    const int* __restrict__ tok,    // [B*SEQ]
                                    __bf16* __restrict__ xbf) {     // [B*SEQ][HID]
  long i = (long)blockIdx.x * blockDim.x + threadIdx.x;
  if (i < (long)BATCH * SEQ * HID) {
    long r = i >> 10;
    int  c = (int)(i & (HID - 1));
    xbf[i] = (__bf16)emb[(long)tok[r] * HID + c];
  }
}

// ---------------- bf16 WMMA GEMM: C[M][N] = A[M][1024] * Bw[N][1024]^T + bias ----------------
// Block = 8 waves sharing one 64-column B strip (blockIdx.x = nq), each wave owns one
// 16-row M band (blockIdx.y*8 + wid). B strip is staged into LDS in double-buffered
// 128-K chunks via async global->LDS copies; A streams from global (L2-resident).
// Requires 128 | M and 64 | N, K = 1024.

#define KCHUNK  128
#define NCHUNKS (1024 / KCHUNK)
#define BROWS   64
#define BSTRIDE 136   // padded LDS row stride in bf16 (4-bank skew, 16B aligned)

__global__ __launch_bounds__(256)
void gemm_bf16_wmma_kernel(const __bf16* __restrict__ A,
                           const __bf16* __restrict__ Bw,
                           const float* __restrict__ bias,
                           float* __restrict__ C,
                           int M, int N) {
  __shared__ __align__(16) __bf16 bstage[2][BROWS * BSTRIDE];   // 2 x 17 KB

  const int tid  = threadIdx.x;
  const int lane = tid & 31;
  const int wid  = tid >> 5;                  // 8 waves / block (wave32)
  const int nq   = blockIdx.x;                // 64-column strip
  const int mt   = blockIdx.y * 8 + wid;      // 16-row band per wave
  const int row  = lane & 15;
  const int half = lane >> 4;

  // Staging assignment: thread copies 64B (32 bf16) of one B row per chunk.
  const int sr = tid >> 2;                    // row 0..63 of the strip
  const int sp = (tid & 3) * 32;              // 32-element part within the chunk
  const __bf16* srow = Bw + ((long)(nq * 64 + sr) << 10) + sp;
  __bf16* lrow = &bstage[0][0] + sr * BSTRIDE + sp;
  const int lbuf = BROWS * BSTRIDE;

  const __bf16* arow = A + ((long)(mt * 16 + row) << 10);

  // Prologue: stage chunk 0 into buffer 0
#pragma unroll
  for (int j = 0; j < 4; ++j)
    stage16(srow + j * 8, lrow + j * 8);

  v8f acc0 = {}, acc1 = {}, acc2 = {}, acc3 = {};

  for (int c = 0; c < NCHUNKS; ++c) {
    const int buf = c & 1;
    wait_async0();        // own chunk-c copies landed
    __syncthreads();      // everyone's chunk c visible; other buffer no longer read

    if (c + 1 < NCHUNKS) {
      const __bf16* s2 = srow + (c + 1) * KCHUNK;
      __bf16* l2 = lrow + ((c + 1) & 1) * lbuf;
#pragma unroll
      for (int j = 0; j < 4; ++j)
        stage16(s2 + j * 8, l2 + j * 8);
    }

    const __bf16* bbase = &bstage[buf][0];
#pragma unroll
    for (int ks = 0; ks < KCHUNK / 32; ++ks) {
      const int kb = c * KCHUNK + ks * 32;
      // A fragment (16x32): row M=lane%16, K = kb + half*16 + [0,16)
      Frag16 a;
      const __bf16* ap = arow + kb + half * 16;
      a.q[0] = *(const uint4*)(ap);
      a.q[1] = *(const uint4*)(ap + 8);
      // B fragments from LDS: col N=tile*16+lane%16, K = kb + half*16 + [0,16)
      const int klocal = ks * 32 + half * 16;
      Frag16 f0, f1, f2, f3;
      const __bf16* p0 = bbase + (0 * 16 + row) * BSTRIDE + klocal;
      const __bf16* p1 = bbase + (1 * 16 + row) * BSTRIDE + klocal;
      const __bf16* p2 = bbase + (2 * 16 + row) * BSTRIDE + klocal;
      const __bf16* p3 = bbase + (3 * 16 + row) * BSTRIDE + klocal;
      f0.q[0] = *(const uint4*)(p0); f0.q[1] = *(const uint4*)(p0 + 8);
      f1.q[0] = *(const uint4*)(p1); f1.q[1] = *(const uint4*)(p1 + 8);
      f2.q[0] = *(const uint4*)(p2); f2.q[1] = *(const uint4*)(p2 + 8);
      f3.q[0] = *(const uint4*)(p3); f3.q[1] = *(const uint4*)(p3 + 8);
      acc0 = wmma_bf16(a.v, f0.v, acc0);
      acc1 = wmma_bf16(a.v, f1.v, acc1);
      acc2 = wmma_bf16(a.v, f2.v, acc2);
      acc3 = wmma_bf16(a.v, f3.v, acc3);
    }
    __syncthreads();      // done reading buf before it is restaged at c+2
  }

  // C/D layout: element (vgpr v, lane): M = mt*16 + v + 8*half, N = tile_base + lane%16
  const int mrow = mt * 16 + 8 * half;
#pragma unroll
  for (int t = 0; t < 4; ++t) {
    v8f acc = (t == 0) ? acc0 : (t == 1) ? acc1 : (t == 2) ? acc2 : acc3;
    const int n = nq * 64 + t * 16 + row;
    const float bv = bias[n];
#pragma unroll
    for (int v = 0; v < 8; ++v)
      C[(long)(mrow + v) * N + n] = acc[v] + bv;
  }
}

// ---------------- GRU recurrence: one workgroup, LDS-resident hidden state ----------------
// xg:  [B][SEQ][3*HID] f32 (precomputed x@w_ih^T + b_ih, gate order r,z,n)
// h0l: [B][HID] f32; Whh: [3*HID][HID] bf16; bhh: [3*HID] f32
// y:   [B][SEQ][HID] bf16 output (next layer's input / logits input)

__global__ __launch_bounds__(1024)
void gru_scan_kernel(const float* __restrict__ xg,
                     const float* __restrict__ h0l,
                     const __bf16* __restrict__ Whh,
                     const float* __restrict__ bhh,
                     __bf16* __restrict__ y) {
  __shared__ __align__(16) float  hS [BATCH * HID];   // 32 KB
  __shared__ __align__(16) __bf16 hbS[BATCH * HID];   // 16 KB
  const int tid  = threadIdx.x;
  const int lane = tid & 31;
  const int wid  = tid >> 5;                 // 32 waves
  const int row  = lane & 15;
  const int half = lane >> 4;

  for (int i = tid; i < BATCH * HID; i += 1024) hS[i] = h0l[i];
  __syncthreads();

  for (int s = 0; s < SEQ; ++s) {
    // h (f32) -> hbS (bf16) for the WMMA A operand
#pragma unroll
    for (int i = 0; i < 8; ++i)
      hbS[i * 1024 + tid] = (__bf16)hS[i * 1024 + tid];
    __syncthreads();

    // 64 column-groups of 16; each wave owns groups {wid, wid+32}, each with r/z/n tiles
    for (int g = wid; g < 64; g += 32) {
      v8f aR = {}, aZ = {}, aN = {};
      const int ncol = g * 16 + row;
      const __bf16* pr = Whh + ((long)(ncol)           << 10);
      const __bf16* pz = Whh + ((long)(HID   + ncol)   << 10);
      const __bf16* pn = Whh + ((long)(2*HID + ncol)   << 10);
      for (int kb = 0; kb < 1024; kb += 32) {
        Frag16 a;
        if (row < 8) {                        // batch rows 0..7 real, 8..15 zero
          const __bf16* ap = hbS + row * 1024 + kb + half * 16;
          a.q[0] = *(const uint4*)(ap);
          a.q[1] = *(const uint4*)(ap + 8);
        } else {
          a.q[0] = make_uint4(0u, 0u, 0u, 0u);
          a.q[1] = make_uint4(0u, 0u, 0u, 0u);
        }
        const int klo = kb + half * 16;
        Frag16 fR, fZ, fN;
        fR.q[0] = *(const uint4*)(pr + klo); fR.q[1] = *(const uint4*)(pr + klo + 8);
        fZ.q[0] = *(const uint4*)(pz + klo); fZ.q[1] = *(const uint4*)(pz + klo + 8);
        fN.q[0] = *(const uint4*)(pn + klo); fN.q[1] = *(const uint4*)(pn + klo + 8);
        aR = wmma_bf16(a.v, fR.v, aR);
        aZ = wmma_bf16(a.v, fZ.v, aZ);
        aN = wmma_bf16(a.v, fN.v, aN);
      }
      // Gate math: valid rows are M=0..7 -> half==0 lanes, vgpr v = batch b
      if (half == 0) {
        const int j = g * 16 + row;
        const float bR = bhh[j], bZ = bhh[HID + j], bN = bhh[2 * HID + j];
#pragma unroll
        for (int v = 0; v < 8; ++v) {
          const float* xrow = xg + ((long)v * SEQ + s) * (3 * HID);
          if (s + 1 < SEQ) __builtin_prefetch(xrow + 3 * HID + j, 0, 0);
          float r  = fast_sigmoid(xrow[j]          + aR[v] + bR);
          float z  = fast_sigmoid(xrow[HID + j]    + aZ[v] + bZ);
          float nn = fast_tanh  (xrow[2 * HID + j] + r * (aN[v] + bN));
          float hp = hS[v * 1024 + j];
          float hn = (1.0f - z) * nn + z * hp;
          hS[v * 1024 + j] = hn;
          y[((long)v * SEQ + s) * HID + j] = (__bf16)hn;
        }
      }
    }
    __syncthreads();   // all h updates visible before next step's conversion
  }
}

// ---------------- launch ----------------

extern "C" void kernel_launch(void* const* d_in, const int* in_sizes, int n_in,
                              void* d_out, int out_size, void* d_ws, size_t ws_size,
                              hipStream_t stream) {
  const float* concepts = (const float*)d_in[0];
  const int*   tokens   = (const int*)  d_in[1];
  const float* emb      = (const float*)d_in[2];
  const float* w_ih     = (const float*)d_in[3];   // [L][3H][H]
  const float* w_hh     = (const float*)d_in[4];   // [L][3H][H]
  const float* b_ih     = (const float*)d_in[5];   // [L][3H]
  const float* b_hh     = (const float*)d_in[6];   // [L][3H]
  const float* fc_w     = (const float*)d_in[7];   // [V][H]
  const float* fc_b     = (const float*)d_in[8];   // [V]
  const float* c2h_w    = (const float*)d_in[9];   // [2H][768]
  const float* c2h_b    = (const float*)d_in[10];  // [2H]
  float* logits = (float*)d_out;                   // [B][S][V] f32

  // Workspace carve-up (all 256B aligned by construction)
  char* ws = (char*)d_ws;
  __bf16* xbf    = (__bf16*)ws;  ws += (long)BATCH * SEQ * HID * 2;          //  8.4 MB
  __bf16* wih_bf = (__bf16*)ws;  ws += (long)LAYERS * 3 * HID * HID * 2;     // 12.6 MB
  __bf16* whh_bf = (__bf16*)ws;  ws += (long)LAYERS * 3 * HID * HID * 2;     // 12.6 MB
  __bf16* fcw_bf = (__bf16*)ws;  ws += (long)VOCAB * HID * 2;                // 65.5 MB
  float*  xg     = (float*)ws;   ws += (long)BATCH * SEQ * 3 * HID * 4;      // 50.3 MB
  float*  gc     = (float*)ws;   ws += (long)BATCH * IN_DIM * 4;
  float*  h0     = (float*)ws;   ws += (long)LAYERS * BATCH * HID * 4;

  const long n_wih = (long)LAYERS * 3 * HID * HID;   // 6.29M
  const long n_fcw = (long)VOCAB * HID;              // 32.77M
  cvt_f32_bf16_kernel<<<(unsigned)((n_wih + 255) / 256), 256, 0, stream>>>(w_ih, wih_bf, n_wih);
  cvt_f32_bf16_kernel<<<(unsigned)((n_wih + 255) / 256), 256, 0, stream>>>(w_hh, whh_bf, n_wih);
  cvt_f32_bf16_kernel<<<(unsigned)((n_fcw + 255) / 256), 256, 0, stream>>>(fc_w, fcw_bf, n_fcw);

  gc_mean_kernel<<<(BATCH * IN_DIM + 255) / 256, 256, 0, stream>>>(concepts, gc);
  h0_kernel<<<(LAYERS * BATCH * HID + 255) / 256, 256, 0, stream>>>(gc, c2h_w, c2h_b, h0);

  const long n_x = (long)BATCH * SEQ * HID;          // 4.19M
  embed_gather_kernel<<<(unsigned)((n_x + 255) / 256), 256, 0, stream>>>(emb, tokens, xbf);

  const int M = BATCH * SEQ;                          // 4096
  {
    dim3 grid_xg(3 * HID / 64, M / 128);              // (48, 32)
    for (int l = 0; l < LAYERS; ++l) {
      gemm_bf16_wmma_kernel<<<grid_xg, 256, 0, stream>>>(
          xbf, wih_bf + (long)l * 3 * HID * HID, b_ih + l * 3 * HID, xg, M, 3 * HID);
      gru_scan_kernel<<<1, 1024, 0, stream>>>(
          xg, h0 + (long)l * BATCH * HID,
          whh_bf + (long)l * 3 * HID * HID, b_hh + l * 3 * HID, xbf);
    }
  }
  {
    dim3 grid_lg(VOCAB / 64, M / 128);                // (500, 32)
    gemm_bf16_wmma_kernel<<<grid_lg, 256, 0, stream>>>(
        xbf, fcw_bf, fc_b, logits, M, VOCAB);
  }
}
